// NeighborAggregator_5291399708711
// MI455X (gfx1250) — matrix operand order repaired and checked by reading, not compiled, not run on hardware
//
#include <hip/hip_runtime.h>

#define NN   16384
#define DIN  128
#define CHUNKS 512          // K chunks of 32
#define KSPLIT 2
#define CPS (CHUNKS / KSPLIT)

typedef __bf16 v16bf __attribute__((ext_vector_type(16)));
typedef float  v8f   __attribute__((ext_vector_type(8)));

// ---------------------------------------------------------------------------
// Kernel 1: pack x[16384][128] (f32, row-major, K-major for the big GEMM) into
// the v_wmma_f32_16x16x32_bf16 B-operand lane layout, chunked by K=32.
// Unit tid = chunk*256 + j*32 + lane; each unit is one lane's 16 bf16 values.
// B order used here (A uses the same K): col = lane&15, K = (lane&16 ? 16 : 0) + e  for element e.
// ---------------------------------------------------------------------------
__global__ __launch_bounds__(256) void k_pack(const float* __restrict__ x,
                                              __bf16* __restrict__ Bbuf) {
  const int tid = blockIdx.x * 256 + threadIdx.x;   // 0 .. 131071
  const int c   = tid >> 8;
  const int rem = tid & 255;
  const int l   = rem & 31;
  const int n   = ((rem >> 5) << 4) + (l & 15);     // j*16 + lane%16
  const int k0  = c * 32 + ((l & 16) ? 16 : 0);
  v16bf o;
#pragma unroll
  for (int e = 0; e < 16; ++e)
    o[e] = (__bf16)x[(size_t)(k0 + e) * DIN + n];
  ((v16bf*)Bbuf)[tid] = o;
}

// ---------------------------------------------------------------------------
// 32-byte L2->LDS async copy (CDNA5 GLOBAL_LOAD_ASYNC_TO_LDS, ASYNCcnt).
// The immediate offset is applied to BOTH the LDS and global addresses, so one
// address pair covers both 16B halves.
// ---------------------------------------------------------------------------
__device__ __forceinline__ void async_copy32(const __bf16* gsrc, unsigned ldst) {
  unsigned long long g = (unsigned long long)gsrc;
  asm volatile("global_load_async_to_lds_b128 %0, %1, off\n\t"
               "global_load_async_to_lds_b128 %0, %1, off offset:16"
               :: "v"(ldst), "v"(g) : "memory");
}
__device__ __forceinline__ void wait_async0() {
  asm volatile("s_wait_asynccnt 0" ::: "memory");
}

// ---------------------------------------------------------------------------
// Kernel 2: nspart[ks] = adj[:, ksplit-slice] @ x_slice  (bf16 WMMA, f32 acc)
//           dpart[ks]  = row-sums of the same adj slice (free from A loads)
// grid = 128 M-blocks * KSPLIT, block = 256 (8 waves, 16 rows/wave).
// B chunks are staged into a 2x8KB LDS double buffer with async-to-LDS copies
// overlapped against the current chunk's WMMAs.
// ---------------------------------------------------------------------------
__global__ __launch_bounds__(256) void k_adj_gemm(const float* __restrict__ adj,
                                                  const __bf16* __restrict__ Bbuf,
                                                  float* __restrict__ nspart,
                                                  float* __restrict__ dpart) {
  __shared__ __bf16 lbuf[2 * 256 * 16];             // 2 x 8KB double buffer
  const int tid  = threadIdx.x;
  const int wave = tid >> 5;
  const int lane = tid & 31;
  const int mblk = blockIdx.x & 127;
  const int ks   = blockIdx.x >> 7;
  const int c0   = ks * CPS;
  const int row  = mblk * 128 + wave * 16 + (lane & 15);
  const int kb   = (lane & 16) ? 16 : 0;            // A-layout K sub-offset

  v8f acc[8] = {};
  float rsum = 0.f;

  const float4* adjv = (const float4*)(adj + (size_t)row * NN);
  v16bf* lb0 = (v16bf*)lbuf;
  // LDS byte address of this thread's 32B staging slot (generic ptr low bits)
  const unsigned lbase = (unsigned)(unsigned long long)(&lbuf[0]) + tid * 32;

  // prologue: async-copy chunk c0 -> buf0
  async_copy32(Bbuf + ((size_t)c0 << 12) + tid * 16, lbase);
  wait_async0();
  __syncthreads();

  for (int cc = 0; cc < CPS; ++cc) {
    const int cur = cc & 1;
    if (cc + 1 < CPS)                               // overlap next-chunk copy
      async_copy32(Bbuf + ((size_t)(c0 + cc + 1) << 12) + tid * 16,
                   lbase + (cur ^ 1) * 8192);

    // --- A tile: 16 f32 of this lane's adj row, f32 -> bf16 ---
    const int k0 = (c0 + cc) * 32;
    const int i0 = (k0 + kb) >> 2;
    float4 a0 = adjv[i0];
    float4 a1 = adjv[i0 + 1];
    float4 a2 = adjv[i0 + 2];                       // K = k0+kb+8
    float4 a3 = adjv[i0 + 3];
    __builtin_prefetch(&adjv[i0 + 64], 0, 1);       // ~8 chunks ahead

    rsum += a0.x + a0.y + a0.z + a0.w + a1.x + a1.y + a1.z + a1.w
          + a2.x + a2.y + a2.z + a2.w + a3.x + a3.y + a3.z + a3.w;

    v16bf av;
    av[0]=(__bf16)a0.x;  av[1]=(__bf16)a0.y;  av[2]=(__bf16)a0.z;  av[3]=(__bf16)a0.w;
    av[4]=(__bf16)a1.x;  av[5]=(__bf16)a1.y;  av[6]=(__bf16)a1.z;  av[7]=(__bf16)a1.w;
    av[8]=(__bf16)a2.x;  av[9]=(__bf16)a2.y;  av[10]=(__bf16)a2.z; av[11]=(__bf16)a2.w;
    av[12]=(__bf16)a3.x; av[13]=(__bf16)a3.y; av[14]=(__bf16)a3.z; av[15]=(__bf16)a3.w;

    const v16bf* lb = lb0 + cur * 256;
#pragma unroll
    for (int j = 0; j < 8; ++j) {
      v16bf bv = lb[j * 32 + lane];
      acc[j] = __builtin_amdgcn_wmma_f32_16x16x32_bf16(
          false, av, false, bv, (short)0, acc[j], false, false);
    }

    wait_async0();                                  // next buffer landed
    __syncthreads();
  }

  // degrees: lane and lane^16 together cover all 32 K per chunk
  rsum += __shfl_xor(rsum, 16, 32);
  if (lane < 16) dpart[(size_t)ks * NN + row] = rsum;

  // store accumulators: C tile layout M = v + (lane&16?8:0), N = lane&15
  float* outp = nspart + ((size_t)ks * NN + (size_t)mblk * 128 + (size_t)wave * 16) * DIN;
  const int n  = lane & 15;
  const int mh = (lane & 16) ? 8 : 0;
#pragma unroll
  for (int j = 0; j < 8; ++j)
#pragma unroll
    for (int v = 0; v < 8; ++v)
      outp[(size_t)(v + mh) * DIN + j * 16 + n] = acc[j][v];
}

// ---------------------------------------------------------------------------
// Kernel 3: one wave per row. Combine K-split partials, divide by clamped
// degree, then self/neigh/comb linears via shfl-broadcast, LayerNorm, ReLU.
// ---------------------------------------------------------------------------
__global__ __launch_bounds__(256) void k_tail(
    const float* __restrict__ x, const float* __restrict__ nspart,
    const float* __restrict__ dpart, const float* __restrict__ self_W,
    const float* __restrict__ self_b, const float* __restrict__ neigh_W,
    const float* __restrict__ neigh_b, const float* __restrict__ comb_W,
    const float* __restrict__ comb_b, const float* __restrict__ gamma,
    const float* __restrict__ beta, float* __restrict__ out) {
  const int lane = threadIdx.x & 31;
  const int row  = blockIdx.x * 8 + (threadIdx.x >> 5);

  const float deg = dpart[row] + dpart[NN + row];
  const float inv = 1.0f / fmaxf(deg, 1.0f);

  float xv[4], nmv[4], s[4], t[4], h[4];
#pragma unroll
  for (int i = 0; i < 4; ++i) {
    const int n = lane + 32 * i;
    xv[i]  = x[(size_t)row * DIN + n];
    nmv[i] = (nspart[(size_t)row * DIN + n] +
              nspart[(size_t)(NN + row) * DIN + n]) * inv;
    s[i] = self_b[n];
    t[i] = neigh_b[n];
    h[i] = comb_b[n];
  }
  // self_feat & neigh_feat
#pragma unroll
  for (int kb = 0; kb < 4; ++kb) {
    for (int u = 0; u < 32; ++u) {
      const int k = kb * 32 + u;
      const float xk = __shfl(xv[kb], u, 32);
      const float mk = __shfl(nmv[kb], u, 32);
      const float* ws = self_W  + (size_t)k * DIN;
      const float* wn = neigh_W + (size_t)k * DIN;
#pragma unroll
      for (int i = 0; i < 4; ++i) {
        s[i] = fmaf(xk, ws[lane + 32 * i], s[i]);
        t[i] = fmaf(mk, wn[lane + 32 * i], t[i]);
      }
    }
  }
  // h = concat(self, neigh) @ comb_W + comb_b
#pragma unroll
  for (int kb = 0; kb < 4; ++kb) {
    for (int u = 0; u < 32; ++u) {
      const int k = kb * 32 + u;
      const float a = __shfl(s[kb], u, 32);
      const float b = __shfl(t[kb], u, 32);
      const float* w0 = comb_W + (size_t)k * DIN;
      const float* w1 = comb_W + (size_t)(k + 128) * DIN;
#pragma unroll
      for (int i = 0; i < 4; ++i)
        h[i] = fmaf(a, w0[lane + 32 * i], fmaf(b, w1[lane + 32 * i], h[i]));
    }
  }
  // LayerNorm over 128 (4 vals/lane x 32 lanes) + ReLU
  float sum = h[0] + h[1] + h[2] + h[3];
  float sq  = h[0]*h[0] + h[1]*h[1] + h[2]*h[2] + h[3]*h[3];
#pragma unroll
  for (int off = 16; off > 0; off >>= 1) {
    sum += __shfl_xor(sum, off, 32);
    sq  += __shfl_xor(sq,  off, 32);
  }
  const float mu  = sum * (1.0f / 128.0f);
  const float var = sq  * (1.0f / 128.0f) - mu * mu;
  const float rs  = rsqrtf(var + 1e-5f);
#pragma unroll
  for (int i = 0; i < 4; ++i) {
    const int n = lane + 32 * i;
    const float y = (h[i] - mu) * rs * gamma[n] + beta[n];
    out[(size_t)row * DIN + n] = fmaxf(y, 0.0f);
  }
}

// ---------------------------------------------------------------------------
extern "C" void kernel_launch(void* const* d_in, const int* in_sizes, int n_in,
                              void* d_out, int out_size, void* d_ws, size_t ws_size,
                              hipStream_t stream) {
  const float* x       = (const float*)d_in[0];
  const float* adj     = (const float*)d_in[1];
  const float* self_W  = (const float*)d_in[2];
  const float* self_b  = (const float*)d_in[3];
  const float* neigh_W = (const float*)d_in[4];
  const float* neigh_b = (const float*)d_in[5];
  const float* comb_W  = (const float*)d_in[6];
  const float* comb_b  = (const float*)d_in[7];
  const float* gamma   = (const float*)d_in[8];
  const float* beta    = (const float*)d_in[9];
  float* out = (float*)d_out;

  char* ws = (char*)d_ws;
  __bf16* Bbuf   = (__bf16*)ws;                                        // 4 MB
  float*  nspart = (float*)(ws + (size_t)NN * DIN * sizeof(__bf16));   // 16 MB
  float*  dpart  = (float*)(ws + (size_t)NN * DIN * sizeof(__bf16)
                               + (size_t)KSPLIT * NN * DIN * sizeof(float)); // 128 KB

  k_pack<<<(NN * DIN) / (256 * 16), 256, 0, stream>>>(x, Bbuf);   // 512 blocks
  k_adj_gemm<<<128 * KSPLIT, 256, 0, stream>>>(adj, Bbuf, nspart, dpart);
  k_tail<<<NN / 8, 256, 0, stream>>>(x, nspart, dpart, self_W, self_b,
                                     neigh_W, neigh_b, comb_W, comb_b,
                                     gamma, beta, out);
}
